// TriangleAttentionStart_84430467105273
// MI455X (gfx1250) — matrix operand order repaired and checked by reading, not compiled, not run on hardware
//
#include <hip/hip_runtime.h>
#include <hip/hip_bf16.h>
#include <math.h>

// ---------------------------------------------------------------------------
// Triangle attention (starting node), MI455X / gfx1250, wave32 + WMMA bf16.
//   C_Z=128, H=W=256, NH=4, C=64, NPIX=65536, NELEM=8388608
// All GEMM-shaped work goes through v_wmma_f32_16x16x32_bf16 (f32 accum).
// Intermediates kept bf16 so they stay resident in the 192MB L2; the final
// f32 output and once-read LN gamma/beta use non-temporal hints so they
// don't evict the working set.
// ---------------------------------------------------------------------------

typedef __bf16 bf16_t;
typedef __attribute__((ext_vector_type(16))) __bf16 v16bf;
typedef __attribute__((ext_vector_type(8)))  float  v8f;

#define NPIX   65536
#define NELEM  8388608
#define LN_EPS 1e-5f

union FragBF { uint4 q[2]; v16bf v; };

__device__ inline v8f wmma_bf16(v16bf a, v16bf b, v8f c) {
  // 8 args: (neg_a, A, neg_b, B, c_mod, C, reuse_a, reuse_b)
  return __builtin_amdgcn_wmma_f32_16x16x32_bf16(false, a, false, b, (short)0, c,
                                                 false, false);
}

// A-fragment (16x32 bf16): lane&15 = M row; lanes<16 hold K {0..7,16..23},
// lanes>=16 hold K {8..15,24..31}.  Two contiguous 16B loads per lane.
__device__ inline v16bf load_frag_a(const bf16_t* rowptr, int kb /*0 or 8*/) {
  FragBF f;
  f.q[0] = *(const uint4*)(rowptr + kb);
  f.q[1] = *(const uint4*)(rowptr + kb + 16);
  return f.v;
}

// B-fragment (32x16 bf16, stored N-major/K-contiguous): lane&15 = N col;
// lanes<16 hold K {0..7,16..23}, lanes>=16 hold K {8..15,24..31}.  Two 16B loads.
__device__ inline v16bf load_frag_b(const bf16_t* rowptr, int kbb /*0 or 16*/) {
  FragBF f;
  f.q[0] = *(const uint4*)(rowptr + (kbb >> 1));
  f.q[1] = *(const uint4*)(rowptr + (kbb >> 1) + 16);
  return f.v;
}

// ---------------------------------------------------------------------------
// Pass 1/2: deterministic two-stage global mean / sum-of-squares reduction.
// ---------------------------------------------------------------------------
__global__ __launch_bounds__(256)
void reduce_partial_kernel(const float* __restrict__ x, float* __restrict__ part) {
  __shared__ float ls[256], ls2[256];
  float s = 0.f, s2 = 0.f;
  size_t base = (size_t)blockIdx.x * 2048 + threadIdx.x;
#pragma unroll
  for (int it = 0; it < 8; ++it) {
    float v = x[base + (size_t)it * 256];
    s += v; s2 += v * v;
  }
  ls[threadIdx.x] = s; ls2[threadIdx.x] = s2;
  __syncthreads();
  for (int off = 128; off > 0; off >>= 1) {
    if ((int)threadIdx.x < off) {
      ls[threadIdx.x]  += ls[threadIdx.x + off];
      ls2[threadIdx.x] += ls2[threadIdx.x + off];
    }
    __syncthreads();
  }
  if (threadIdx.x == 0) {
    part[blockIdx.x * 2 + 0] = ls[0];
    part[blockIdx.x * 2 + 1] = ls2[0];
  }
}

__global__ __launch_bounds__(256)
void reduce_final_kernel(const float* __restrict__ part, float* __restrict__ red) {
  __shared__ float ls[256], ls2[256];
  float s = 0.f, s2 = 0.f;
  for (int i = threadIdx.x; i < 4096; i += 256) {
    s += part[i * 2 + 0]; s2 += part[i * 2 + 1];
  }
  ls[threadIdx.x] = s; ls2[threadIdx.x] = s2;
  __syncthreads();
  for (int off = 128; off > 0; off >>= 1) {
    if ((int)threadIdx.x < off) {
      ls[threadIdx.x]  += ls[threadIdx.x + off];
      ls2[threadIdx.x] += ls2[threadIdx.x + off];
    }
    __syncthreads();
  }
  if (threadIdx.x == 0) {
    float mean = ls[0] * (1.0f / (float)NELEM);
    float var  = ls2[0] * (1.0f / (float)NELEM) - mean * mean;
    red[0] = mean;
    red[1] = rsqrtf(var + LN_EPS);
  }
}

// ---------------------------------------------------------------------------
// fp32 -> bf16 weight conversion
// ---------------------------------------------------------------------------
__global__ void cvt_bf16_kernel(const float* __restrict__ src,
                                bf16_t* __restrict__ dst, int n) {
  int i = blockIdx.x * blockDim.x + threadIdx.x;
  if (i < n) dst[i] = (bf16_t)src[i];
}

// ---------------------------------------------------------------------------
// LayerNorm + transpose:  x[c][p] -> xt[p][c]  (bf16), via 32x32 LDS tile.
// gamma/beta are read exactly once -> non-temporal loads.
// ---------------------------------------------------------------------------
__global__ __launch_bounds__(256)
void ln_transpose_kernel(const float* __restrict__ x, const float* __restrict__ lw,
                         const float* __restrict__ lb, const float* __restrict__ red,
                         bf16_t* __restrict__ xt) {
  __shared__ float tile[32][33];
  const float mu = red[0];
  const float rs = red[1];
  const int p0 = blockIdx.x * 32;
  const int c0 = blockIdx.y * 32;
  const int tx = threadIdx.x;    // 0..31
  const int ty = threadIdx.y;    // 0..7
#pragma unroll
  for (int yy = 0; yy < 32; yy += 8) {
    const int c = c0 + ty + yy;
    const size_t idx = (size_t)c * NPIX + (p0 + tx);
    const float g = __builtin_nontemporal_load(&lw[idx]);
    const float b = __builtin_nontemporal_load(&lb[idx]);
    tile[ty + yy][tx] = (x[idx] - mu) * rs * g + b;
  }
  __syncthreads();
#pragma unroll
  for (int yy = 0; yy < 32; yy += 8) {
    const int p = p0 + ty + yy;
    xt[(size_t)p * 128 + (c0 + tx)] = (bf16_t)tile[tx][ty + yy];
  }
}

// ---------------------------------------------------------------------------
// NT GEMM: D[M,N] = A[M,K] * B[N,K]^T, A/B bf16 K-contiguous, f32 accum.
// Block = 256 thr (8 waves), block tile 64(M) x 128(N).
// Each wave owns a 16(M) x 64(N) strip: one A fragment feeds 4 WMMAs per
// K-step (2.5 vmem loads per WMMA instead of 4).
// Epilogues: 0 = f32 out + bias, non-temporal (final projection -> d_out)
//            1 = bf16 scatter   [h][p][c]          (q, k)
//            2 = sigmoid(.+b)   [h][p][c]          (gate)
//            3 = bf16 scatter   [h][i][c][j] (V^T) (v)
// ---------------------------------------------------------------------------
template <int EPI>
__global__ __launch_bounds__(256)
void gemm_nt_kernel(const bf16_t* __restrict__ A, const bf16_t* __restrict__ B,
                    int M, int N, int K, float* __restrict__ outF,
                    bf16_t* __restrict__ outB, const float* __restrict__ biasv) {
  const int lane  = threadIdx.x & 31;
  const int wv    = threadIdx.x >> 5;
  const int l15   = lane & 15;
  const int kb    = (lane & 16) ? 8 : 0;
  const int kbb   = (lane & 16) ? 16 : 0;
  const int rhalf = (lane & 16) ? 8 : 0;

  const int m0 = blockIdx.y * 64 + (wv >> 1) * 16;   // wave M row block
  const int nb = blockIdx.x * 128 + (wv & 1) * 64;   // wave N strip (4 subtiles)

  const bf16_t* arow = A + (size_t)(m0 + l15) * K;
  const bf16_t* brow = B + (size_t)(nb + l15) * K;

  v8f acc[4] = {};
  for (int k0 = 0; k0 < K; k0 += 32) {
    if (k0 + 32 < K) __builtin_prefetch(brow + k0 + 32, 0, 0);
    const v16bf fa = load_frag_a(arow + k0, kb);
#pragma unroll
    for (int nt = 0; nt < 4; ++nt) {
      const v16bf fb = load_frag_b(brow + (size_t)nt * 16 * K + k0, kbb);
      acc[nt] = wmma_bf16(fa, fb, acc[nt]);
    }
  }

#pragma unroll
  for (int nt = 0; nt < 4; ++nt) {
    const int n = nb + nt * 16 + l15;
#pragma unroll
    for (int r = 0; r < 8; ++r) {
      const int m = m0 + r + rhalf;
      const float v = acc[nt][r];
      if (EPI == 0) {
        __builtin_nontemporal_store(v + biasv[m], &outF[(size_t)m * N + n]);
      } else if (EPI == 1) {
        const int h = m >> 6, c = m & 63;
        outB[((size_t)(h * NPIX + n)) * 64 + c] = (bf16_t)v;
      } else if (EPI == 2) {
        const int h = m >> 6, c = m & 63;
        const float g = 1.0f / (1.0f + __expf(-(v + biasv[m])));
        outB[((size_t)(h * NPIX + n)) * 64 + c] = (bf16_t)g;
      } else {  // EPI == 3 : V^T layout [h][i][c][j]
        const int h = m >> 6, c = m & 63;
        const int i = n >> 8, j = n & 255;
        outB[((size_t)(h * 16384 + i * 64 + c)) * 256 + j] = (bf16_t)v;
      }
    }
  }
}

// ---------------------------------------------------------------------------
// bias[h][p] = sum_c wb[h][c] * x[c][p]   (tiny: 33M MACs, plain VALU)
// ---------------------------------------------------------------------------
__global__ __launch_bounds__(256)
void bias_proj_kernel(const bf16_t* __restrict__ xt, const float* __restrict__ wb,
                      float* __restrict__ biasb) {
  const int p = blockIdx.x * blockDim.x + threadIdx.x;
  const bf16_t* xr = xt + (size_t)p * 128;
  float acc[4] = {0.f, 0.f, 0.f, 0.f};
  for (int c = 0; c < 128; ++c) {
    const float xv = (float)xr[c];
#pragma unroll
    for (int hh = 0; hh < 4; ++hh) acc[hh] += wb[hh * 128 + c] * xv;
  }
#pragma unroll
  for (int hh = 0; hh < 4; ++hh) biasb[(size_t)hh * NPIX + p] = acc[hh];
}

// ---------------------------------------------------------------------------
// Fused attention per (h, i, 32-row j panel):
//   S = scale*Q K^T + bias[h,j,k]  (WMMA)  -> LDS f32 [32][256]
//   softmax over k (wave shfl)             -> LDS bf16 [32][256]
//   O = A V  (WMMA), gate, scatter to og[p][c*4+h] (merge order (c,h))
// Phase 1: each wave keeps its two Q fragments in registers and sweeps
// four 16-wide K-column tiles (Q loads amortized 4x).
// ---------------------------------------------------------------------------
__global__ __launch_bounds__(256)
void tri_attention_kernel(const bf16_t* __restrict__ q_s, const bf16_t* __restrict__ k_s,
                          const bf16_t* __restrict__ vt_s, const bf16_t* __restrict__ g_s,
                          const float* __restrict__ biasb, bf16_t* __restrict__ og) {
  __shared__ float  s_lds[32][256];   // 32 KB
  __shared__ bf16_t a_lds[32][256];   // 16 KB

  const int lane  = threadIdx.x & 31;
  const int wv    = threadIdx.x >> 5;
  const int l15   = lane & 15;
  const int kb    = (lane & 16) ? 8 : 0;
  const int kbb   = (lane & 16) ? 16 : 0;
  const int rhalf = (lane & 16) ? 8 : 0;

  const int j0 = blockIdx.x * 32;
  const int i  = blockIdx.y;
  const int h  = blockIdx.z;
  const float scale = 0.125f;  // 1/sqrt(64)

  // ---- Phase 1: score panel  S[32][256] ----
  {
    const int jt  = wv & 1;        // wave's 16-row j tile
    const int ktg = wv >> 1;       // wave's group of 4 k-column tiles
    const int j   = j0 + jt * 16 + l15;
    const bf16_t* qrow = q_s + ((size_t)(h * NPIX + i * 256 + j)) * 64;
    const v16bf qa0 = load_frag_a(qrow, kb);        // c = 0..31
    const v16bf qa1 = load_frag_a(qrow + 32, kb);   // c = 32..63
#pragma unroll
    for (int kt = ktg * 4; kt < ktg * 4 + 4; ++kt) {
      const int k0col = kt * 16;
      const bf16_t* krow = k_s + ((size_t)(h * NPIX + i * 256 + k0col + l15)) * 64;
      v8f acc = {};
      acc = wmma_bf16(qa0, load_frag_b(krow, kbb), acc);
      acc = wmma_bf16(qa1, load_frag_b(krow + 32, kbb), acc);
#pragma unroll
      for (int r = 0; r < 8; ++r) {
        const int jj = jt * 16 + r + rhalf;
        const int kc = k0col + l15;
        const float bv = biasb[(size_t)h * NPIX + (size_t)(j0 + jj) * 256 + kc];
        s_lds[jj][kc] = acc[r] * scale + bv;
      }
    }
  }
  __syncthreads();

  // ---- Phase 2: softmax over k (8 lanes per row, 32 elems each) ----
  {
    const int row = threadIdx.x >> 3;   // 0..31
    const int seg = threadIdx.x & 7;
    float* srow = s_lds[row];
    float m = -3.0e38f;
#pragma unroll 8
    for (int kk = seg * 32; kk < seg * 32 + 32; ++kk) m = fmaxf(m, srow[kk]);
    m = fmaxf(m, __shfl_xor(m, 1));
    m = fmaxf(m, __shfl_xor(m, 2));
    m = fmaxf(m, __shfl_xor(m, 4));
    float sum = 0.f;
#pragma unroll 8
    for (int kk = seg * 32; kk < seg * 32 + 32; ++kk) {
      const float e = __expf(srow[kk] - m);
      srow[kk] = e;
      sum += e;
    }
    sum += __shfl_xor(sum, 1);
    sum += __shfl_xor(sum, 2);
    sum += __shfl_xor(sum, 4);
    const float inv = 1.0f / sum;
#pragma unroll 8
    for (int kk = seg * 32; kk < seg * 32 + 32; ++kk)
      a_lds[row][kk] = (bf16_t)(srow[kk] * inv);
  }
  __syncthreads();

  // ---- Phase 3: O = A V, gate, scatter (one 16x16 tile per wave) ----
  {
    const int jt = wv >> 2;                 // 0..1
    const int ct = wv & 3;                  // 0..3
    const int jj = jt * 16 + l15;
    const int c  = ct * 16 + l15;
    const bf16_t* arow = &a_lds[jj][0];
    const bf16_t* vrow = vt_s + ((size_t)(h * 16384 + i * 64 + c)) * 256;
    v8f acc = {};
#pragma unroll
    for (int k0 = 0; k0 < 256; k0 += 32) {
      v16bf fa = load_frag_a(arow + k0, kb);   // ds_load_b128 x2
      v16bf fb = load_frag_b(vrow + k0, kbb);  // global_load_b128 x2
      acc = wmma_bf16(fa, fb, acc);
    }
#pragma unroll
    for (int r = 0; r < 8; ++r) {
      const int jrow = jt * 16 + r + rhalf;
      const int j = j0 + jrow;
      const float gv = (float)g_s[((size_t)(h * NPIX + i * 256 + j)) * 64 + c];
      og[((size_t)(i * 256 + j)) * 256 + (c * 4 + h)] = (bf16_t)(acc[r] * gv);
    }
  }
}

// ---------------------------------------------------------------------------
// Host-side orchestration (graph-capture safe: kernels only, all on stream).
// ---------------------------------------------------------------------------
extern "C" void kernel_launch(void* const* d_in, const int* in_sizes, int n_in,
                              void* d_out, int out_size, void* d_ws, size_t ws_size,
                              hipStream_t stream) {
  (void)in_sizes; (void)n_in; (void)out_size; (void)ws_size;
  const float* pair   = (const float*)d_in[0];
  const float* ln_w   = (const float*)d_in[1];
  const float* ln_b   = (const float*)d_in[2];
  const float* wq     = (const float*)d_in[3];
  const float* wk     = (const float*)d_in[4];
  const float* wvf    = (const float*)d_in[5];
  const float* wb     = (const float*)d_in[6];
  const float* wg_w   = (const float*)d_in[7];
  const float* wg_b   = (const float*)d_in[8];
  const float* wrep_w = (const float*)d_in[9];
  const float* wrep_b = (const float*)d_in[10];
  float* outp = (float*)d_out;

  char* ws = (char*)d_ws;
  size_t off = 0;
  auto take = [&](size_t bytes) -> char* {
    char* p = ws + off;
    off = (off + bytes + 255) & ~(size_t)255;
    return p;
  };
  float*  red    = (float*)take(2 * sizeof(float));
  float*  part   = (float*)take(4096 * 2 * sizeof(float));
  bf16_t* xt     = (bf16_t*)take((size_t)NPIX * 128 * 2);      // x^T  [p][c]
  bf16_t* qs     = (bf16_t*)take((size_t)4 * NPIX * 64 * 2);   // q    [h][p][c]
  bf16_t* ks     = (bf16_t*)take((size_t)4 * NPIX * 64 * 2);   // k    [h][p][c]
  bf16_t* vts    = (bf16_t*)take((size_t)4 * NPIX * 64 * 2);   // v^T  [h][i][c][j]
  bf16_t* gs     = (bf16_t*)take((size_t)4 * NPIX * 64 * 2);   // gate [h][p][c]
  bf16_t* og     = (bf16_t*)take((size_t)NPIX * 256 * 2);      // o    [p][c*4+h]
  float*  biasb  = (float*)take((size_t)4 * NPIX * sizeof(float));
  bf16_t* wq_b   = (bf16_t*)take(32768 * 2);
  bf16_t* wk_b   = (bf16_t*)take(32768 * 2);
  bf16_t* wv_b   = (bf16_t*)take(32768 * 2);
  bf16_t* wg_bq  = (bf16_t*)take(32768 * 2);
  bf16_t* wrep_q = (bf16_t*)take(32768 * 2);

  // 1) global mean/var (deterministic two-stage)
  reduce_partial_kernel<<<4096, 256, 0, stream>>>(pair, part);
  reduce_final_kernel<<<1, 256, 0, stream>>>(part, red);

  // 2) weights -> bf16
  cvt_bf16_kernel<<<128, 256, 0, stream>>>(wq, wq_b, 32768);
  cvt_bf16_kernel<<<128, 256, 0, stream>>>(wk, wk_b, 32768);
  cvt_bf16_kernel<<<128, 256, 0, stream>>>(wvf, wv_b, 32768);
  cvt_bf16_kernel<<<128, 256, 0, stream>>>(wg_w, wg_bq, 32768);
  cvt_bf16_kernel<<<128, 256, 0, stream>>>(wrep_w, wrep_q, 32768);

  // 3) LayerNorm + transpose to [p][c] bf16
  ln_transpose_kernel<<<dim3(2048, 4), dim3(32, 8), 0, stream>>>(pair, ln_w, ln_b,
                                                                 red, xt);

  // 4) projections: M=256, N=65536, K=128 WMMA GEMMs
  dim3 gproj(512, 4);
  gemm_nt_kernel<1><<<gproj, 256, 0, stream>>>(wq_b, xt, 256, NPIX, 128, nullptr, qs, nullptr);
  gemm_nt_kernel<1><<<gproj, 256, 0, stream>>>(wk_b, xt, 256, NPIX, 128, nullptr, ks, nullptr);
  gemm_nt_kernel<3><<<gproj, 256, 0, stream>>>(wv_b, xt, 256, NPIX, 128, nullptr, vts, nullptr);
  gemm_nt_kernel<2><<<gproj, 256, 0, stream>>>(wg_bq, xt, 256, NPIX, 128, nullptr, gs, wg_b);

  // 5) pair bias  bias[h][j*256+k]
  bias_proj_kernel<<<256, 256, 0, stream>>>(xt, wb, biasb);

  // 6) fused attention: (jblock=8) x (i=256) x (h=4)
  tri_attention_kernel<<<dim3(8, 256, 4), 256, 0, stream>>>(qs, ks, vts, gs, biasb, og);

  // 7) output projection: M=128, N=65536, K=256, f32 NT out + bias -> d_out
  gemm_nt_kernel<0><<<dim3(512, 2), 256, 0, stream>>>(wrep_q, og, 128, NPIX, 256,
                                                      outp, nullptr, wrep_b);
}
